// SimpleTransformerLM_53669911331227
// MI455X (gfx1250) — compile-verified, matrix operand order checked by reading, not run on hardware
//
#include <hip/hip_runtime.h>
#include <hip/hip_bf16.h>
#include <math.h>

// ---------------- problem constants ----------------
#define L_   2
#define D_   512
#define G_   4
#define E_   2
#define H_   1024
#define V_   50257
#define NH_  8
#define B_   2
#define S_   1024
#define N_   (B_ * S_)
#define NEXP (G_ * E_)
#define EPS_ 1e-5f
#define TN   4      // 16x16 output tiles per wave (A-fragment reuse)

typedef __attribute__((ext_vector_type(16))) __bf16 v16bf;
typedef __attribute__((ext_vector_type(8)))  __bf16 v8bf;
typedef __attribute__((ext_vector_type(8)))  float  v8f;

// fp32 -> bf16, round-to-nearest-even (bit trick)
__device__ __forceinline__ __bf16 f2bf(float x) {
    union { float f; unsigned int u; } a; a.f = x;
    unsigned int r = a.u + 0x7FFFu + ((a.u >> 16) & 1u);
    union { unsigned short s; __bf16 b; } o; o.s = (unsigned short)(r >> 16);
    return o.b;
}

// bf16 fragment load, K-contiguous memory: two 16B vector loads.
// Lane (lane%16) owns row/col; kg = lane/16:
//   elements 0..7  -> K = kg*8 + i ; elements 8..15 -> K = 16 + kg*8 + i
__device__ __forceinline__ v16bf load_frag_bf(const __bf16* __restrict__ p, int kg) {
    union { uint4 q; v8bf h; } u0, u1;
    u0.q = *(const uint4*)(p + kg * 8);
    u1.q = *(const uint4*)(p + 16 + kg * 8);
    return __builtin_shufflevector(u0.h, u1.h, 0,1,2,3,4,5,6,7,8,9,10,11,12,13,14,15);
}

__device__ __forceinline__ void wait_async0() {
#if __has_builtin(__builtin_amdgcn_s_wait_asynccnt)
    __builtin_amdgcn_s_wait_asynccnt(0);
#else
    asm volatile("s_wait_asynccnt 0" ::: "memory");
#endif
}

// ---------------- weight prep: fp32 -> bf16 (optionally transposed) ----------------
__global__ __launch_bounds__(256)
void cvt_bf16_kernel(const float* __restrict__ s, __bf16* __restrict__ d, long n) {
    long t = (long)blockIdx.x * 256 + threadIdx.x;
    if (t < n) d[t] = f2bf(s[t]);
}

// per-matrix transpose: src [R,C] -> dst [C,R]; blockIdx.z selects matrix
__global__ __launch_bounds__(256)
void transpose_cvt_kernel(const float* __restrict__ src, __bf16* __restrict__ dst,
                          int R, int C) {
    const float* s = src + (size_t)blockIdx.z * R * C;
    __bf16*      d = dst + (size_t)blockIdx.z * R * C;
    int t = blockIdx.x * 256 + threadIdx.x;
    if (t >= R * C) return;
    int r = t / C, cc = t - r * C;
    d[(size_t)cc * R + r] = f2bf(s[t]);
}

// ---------------- embedding ----------------
__global__ __launch_bounds__(256)
void embed_kernel(const int* __restrict__ tokens, const float* __restrict__ tok_emb,
                  const float* __restrict__ pos_emb, float* __restrict__ h) {
    int t = blockIdx.x * 256 + threadIdx.x;
    if (t >= N_ * D_) return;
    int n = t / D_, d = t % D_;
    int s = n & (S_ - 1);
    h[t] = tok_emb[(size_t)tokens[n] * D_ + d] + pos_emb[(size_t)s * D_ + d];
}

// ---------------- layernorm (wave per row): writes f32 + bf16 copies ----------------
__global__ __launch_bounds__(256)
void layernorm(const float* __restrict__ x, const float* __restrict__ g,
               const float* __restrict__ b, float* __restrict__ yf,
               __bf16* __restrict__ yb) {
    int wave = threadIdx.x >> 5, lane = threadIdx.x & 31;
    int n = blockIdx.x * 8 + wave;
    if (n >= N_) return;
    const float* xr = x + (size_t)n * D_;
    float s = 0.f, ss = 0.f;
    for (int d = lane; d < D_; d += 32) { float v = xr[d]; s += v; ss += v * v; }
#pragma unroll
    for (int off = 16; off; off >>= 1) { s += __shfl_xor(s, off); ss += __shfl_xor(ss, off); }
    float mean = s * (1.0f / D_);
    float var  = ss * (1.0f / D_) - mean * mean;
    float rs   = rsqrtf(var + EPS_);
    float* yr  = yf + (size_t)n * D_;
    __bf16* ybr = yb + (size_t)n * D_;
    for (int d = lane; d < D_; d += 32) {
        float v = (xr[d] - mean) * rs * g[d] + b[d];
        yr[d] = v;
        ybr[d] = f2bf(v);
    }
}

// ---------------- dense WMMA GEMM, bf16 operands ----------------
// Double-buffered async-LDS A-tile staging; clamped B rows (no divergence in k-loop).
// C[m,n] (=|+=) sum_k A[m,k]*B[n,k] + bias[n];  A,B bf16 row-major K-contiguous.
__global__ __launch_bounds__(256)
void wmma_gemm_bf(const __bf16* __restrict__ A, int lda,
                  const __bf16* __restrict__ B, const float* __restrict__ bias,
                  float* __restrict__ C, int ldc,
                  int M, int Nc, int K, int accum) {
    __shared__ __align__(16) unsigned char sA[2][16 * 64];   // 2 x (16 rows x 32 bf16)
    int wave  = threadIdx.x >> 5;
    int lane  = threadIdx.x & 31;
    int row16 = lane & 15;
    int kg    = lane >> 4;
    int m0    = blockIdx.x * 16;
    int nt0   = (blockIdx.y * 8 + wave) * TN;

    // clamped B row pointers: OOB lanes read row Nc-1 (valid memory), stores guarded
    const __bf16* bp[TN];
    int ncol[TN];
#pragma unroll
    for (int t = 0; t < TN; t++) {
        int n = (nt0 + t) * 16 + row16;
        ncol[t] = n;
        int nc = (n < Nc) ? n : (Nc - 1);
        bp[t] = B + (size_t)nc * K;
    }

    v8f c[TN];
#pragma unroll
    for (int t = 0; t < TN; t++)
#pragma unroll
        for (int r = 0; r < 8; r++) c[t][r] = 0.0f;

    unsigned ldsbase = (unsigned)(uintptr_t)(void*)sA;
    int sr = threadIdx.x >> 2, sch = threadIdx.x & 3;        // staging row / 16B chunk
    const __bf16* asrc = A + (size_t)(m0 + sr) * lda + sch * 8;

    // prologue: stage k0=0 into buffer 0
    if (threadIdx.x < 64) {
        asm volatile("global_load_async_to_lds_b128 %0, %1, off"
                     :: "v"(ldsbase + sr * 64 + sch * 16),
                        "v"((unsigned long long)(uintptr_t)asrc)
                     : "memory");
    }
    unsigned cur = 0;
    for (int k0 = 0; k0 < K; k0 += 32) {
        wait_async0();                                       // staged tile landed in LDS
        __syncthreads();                                     // all waves see it / done with other buf
        if ((k0 + 32 < K) && threadIdx.x < 64) {             // stage next slab into other buffer
            asm volatile("global_load_async_to_lds_b128 %0, %1, off"
                         :: "v"(ldsbase + ((cur ^ 1u) << 10) + sr * 64 + sch * 16),
                            "v"((unsigned long long)(uintptr_t)(asrc + k0 + 32))
                         : "memory");
        }
        const uint4* la = (const uint4*)(sA[cur] + row16 * 64 + kg * 16);
        union { uint4 q; v8bf h; } u0, u1;
        u0.q = la[0]; u1.q = la[2];                          // K lo / K hi halves
        v16bf a = __builtin_shufflevector(u0.h, u1.h,
                   0,1,2,3,4,5,6,7,8,9,10,11,12,13,14,15);
        v16bf bf[TN];
#pragma unroll
        for (int t = 0; t < TN; t++) {
            if (k0 + 64 < K) __builtin_prefetch((const void*)(bp[t] + k0 + 64), 0, 1);
            bf[t] = load_frag_bf(bp[t] + k0, kg);
        }
#pragma unroll
        for (int t = 0; t < TN; t++)
            c[t] = __builtin_amdgcn_wmma_f32_16x16x32_bf16(false, a, false, bf[t],
                                                           (short)0, c[t], false, false);
        cur ^= 1u;
    }
#pragma unroll
    for (int t = 0; t < TN; t++) {
        if (ncol[t] >= Nc) continue;
        float bv = bias[ncol[t]];
#pragma unroll
        for (int r = 0; r < 8; r++) {
            int mrow = m0 + r + 8 * kg;
            size_t o = (size_t)mrow * ldc + ncol[t];
            float val = c[t][r] + bv;
            if (accum) C[o] += val; else C[o] = val;
        }
    }
}

// ---------------- attention: streaming causal softmax, wave per (b,h,q) ----------------
__global__ __launch_bounds__(256)
void attn_kernel(const float* __restrict__ qkv, __bf16* __restrict__ outp) {
    int wave = threadIdx.x >> 5, lane = threadIdx.x & 31;
    int gw = blockIdx.x * 8 + wave;
    int q  = gw & (S_ - 1);
    int hh = (gw >> 10) & (NH_ - 1);
    int b  = gw >> 13;
    const float* qr = qkv + (size_t)(b * S_ + q) * (3 * D_) + hh * 64;
    float q0 = qr[lane], q1 = qr[lane + 32];
    float m = -INFINITY, lsum = 0.f, a0 = 0.f, a1 = 0.f;
    for (int k = 0; k <= q; k++) {
        const float* kr = qkv + (size_t)(b * S_ + k) * (3 * D_) + D_ + hh * 64;
        float s = q0 * kr[lane] + q1 * kr[lane + 32];
#pragma unroll
        for (int off = 16; off; off >>= 1) s += __shfl_xor(s, off);
        s *= 0.125f;                          // 1/sqrt(64)
        float mn = fmaxf(m, s);
        float al = __expf(m - mn);
        float p  = __expf(s - mn);
        const float* vr = qkv + (size_t)(b * S_ + k) * (3 * D_) + 2 * D_ + hh * 64;
        a0 = a0 * al + p * vr[lane];
        a1 = a1 * al + p * vr[lane + 32];
        lsum = lsum * al + p;
        m = mn;
    }
    float inv = 1.0f / lsum;
    __bf16* orow = outp + (size_t)(b * S_ + q) * D_ + hh * 64;
    orow[lane] = f2bf(a0 * inv); orow[lane + 32] = f2bf(a1 * inv);
}

// ---------------- MoE router: top-1 group then top-1 expert; wave per token ----------------
__global__ __launch_bounds__(256)
void moe_router(const float* __restrict__ x,
                const float* __restrict__ grw, const float* __restrict__ grb,
                const float* __restrict__ erw, const float* __restrict__ erb,
                int* __restrict__ counts, int* __restrict__ idx,
                float* __restrict__ gate) {
    int wave = threadIdx.x >> 5, lane = threadIdx.x & 31;
    int n = blockIdx.x * 8 + wave;
    if (n >= N_) return;
    const float* xr = x + (size_t)n * D_;
    float acc[G_] = {0.f, 0.f, 0.f, 0.f};
    for (int d = lane; d < D_; d += 32) {
        float xv = xr[d];
#pragma unroll
        for (int g = 0; g < G_; g++) acc[g] += xv * grw[g * D_ + d];
    }
#pragma unroll
    for (int g = 0; g < G_; g++)
#pragma unroll
        for (int off = 16; off; off >>= 1) acc[g] += __shfl_xor(acc[g], off);
    float gl[G_];
#pragma unroll
    for (int g = 0; g < G_; g++) gl[g] = acc[g] + grb[g];
    int gi = 0; float gm = gl[0];
#pragma unroll
    for (int g = 1; g < G_; g++) if (gl[g] > gm) { gm = gl[g]; gi = g; }
    float se = 0.f;
#pragma unroll
    for (int g = 0; g < G_; g++) se += __expf(gl[g] - gm);
    float gpi = 1.0f / se;

    float ea[E_] = {0.f, 0.f};
    for (int d = lane; d < D_; d += 32) {
        float xv = xr[d];
#pragma unroll
        for (int e = 0; e < E_; e++) ea[e] += xv * erw[(size_t)(gi * E_ + e) * D_ + d];
    }
#pragma unroll
    for (int e = 0; e < E_; e++)
#pragma unroll
        for (int off = 16; off; off >>= 1) ea[e] += __shfl_xor(ea[e], off);
    float el[E_];
#pragma unroll
    for (int e = 0; e < E_; e++) el[e] = ea[e] + erb[gi * E_ + e];
    int ei = 0; float em = el[0];
#pragma unroll
    for (int e = 1; e < E_; e++) if (el[e] > em) { em = el[e]; ei = e; }
    float ses = 0.f;
#pragma unroll
    for (int e = 0; e < E_; e++) ses += __expf(el[e] - em);
    float epi = 1.0f / ses;

    if (lane == 0) {
        int eid = gi * E_ + ei;
        int pos = atomicAdd(&counts[eid], 1);
        idx[(size_t)eid * N_ + pos] = n;
        gate[n] = gpi * epi;
    }
}

// ---------------- gathered WMMA GEMM for routed experts (bf16 operands) ----------------
// mode 0: Cb[tok,n] = bf16(relu(acc + bias[n]))        (FFN1 -> h1 bf16)
// mode 1: Cf[tok,n] += gate[tok] * (acc + bias[n])     (FFN2 -> residual h, f32)
__global__ __launch_bounds__(256)
void moe_gemm_bf(const __bf16* __restrict__ A, int lda,
                 const __bf16* __restrict__ Bbase, long bStride,
                 const float* __restrict__ biasBase, int biasStride,
                 __bf16* __restrict__ Cb, float* __restrict__ Cf, int ldc,
                 const int* __restrict__ counts, const int* __restrict__ idxBase,
                 const float* __restrict__ gate,
                 int Nc, int K, int mode) {
    int eid = blockIdx.z;
    int cnt = counts[eid];
    int m0  = blockIdx.x * 16;
    if (m0 >= cnt) return;
    int wave  = threadIdx.x >> 5;
    int lane  = threadIdx.x & 31;
    int row16 = lane & 15;
    int kg    = lane >> 4;
    int nt0 = (blockIdx.y * 8 + wave) * TN;
    if (nt0 * 16 >= Nc) return;

    const int* idx = idxBase + (size_t)eid * N_;
    int mrowg = m0 + row16;
    int tokA  = (mrowg < cnt) ? idx[mrowg] : idx[cnt - 1];   // clamp to valid token
    const __bf16* arow = A + (size_t)tokA * lda;
    const __bf16* B    = Bbase + (size_t)eid * bStride;
    const float*  bias = biasBase + (size_t)eid * biasStride;

    // clamped B row pointers, hoisted out of k-loop
    const __bf16* bp[TN];
    int ncol[TN];
#pragma unroll
    for (int t = 0; t < TN; t++) {
        int n = (nt0 + t) * 16 + row16;
        ncol[t] = n;
        int nc = (n < Nc) ? n : (Nc - 1);
        bp[t] = B + (size_t)nc * K;
    }

    v8f c[TN];
#pragma unroll
    for (int t = 0; t < TN; t++)
#pragma unroll
        for (int r = 0; r < 8; r++) c[t][r] = 0.0f;

    for (int k0 = 0; k0 < K; k0 += 32) {
        v16bf a = load_frag_bf(arow + k0, kg);
        v16bf bf[TN];
#pragma unroll
        for (int t = 0; t < TN; t++) {
            if (k0 + 64 < K) __builtin_prefetch((const void*)(bp[t] + k0 + 64), 0, 1);
            bf[t] = load_frag_bf(bp[t] + k0, kg);
        }
#pragma unroll
        for (int t = 0; t < TN; t++)
            c[t] = __builtin_amdgcn_wmma_f32_16x16x32_bf16(false, a, false, bf[t],
                                                           (short)0, c[t], false, false);
    }
#pragma unroll
    for (int t = 0; t < TN; t++) {
        if (ncol[t] >= Nc) continue;
        float bv = bias[ncol[t]];
#pragma unroll
        for (int r = 0; r < 8; r++) {
            int mi = m0 + r + 8 * kg;
            if (mi >= cnt) continue;
            int tok = idx[mi];
            float val = c[t][r] + bv;
            if (mode == 0) Cb[(size_t)tok * ldc + ncol[t]] = f2bf(fmaxf(val, 0.0f));
            else           Cf[(size_t)tok * ldc + ncol[t]] += gate[tok] * val;
        }
    }
}

// ---------------- host orchestration ----------------
extern "C" void kernel_launch(void* const* d_in, const int* in_sizes, int n_in,
                              void* d_out, int out_size, void* d_ws, size_t ws_size,
                              hipStream_t stream) {
    (void)in_sizes; (void)n_in; (void)out_size; (void)ws_size;
    const int*   tokens  = (const int*)  d_in[0];
    const float* tok_emb = (const float*)d_in[1];
    const float* pos_emb = (const float*)d_in[2];
    const float* wqkv    = (const float*)d_in[3];
    const float* bqkv    = (const float*)d_in[4];
    const float* wo      = (const float*)d_in[5];
    const float* bo      = (const float*)d_in[6];
    const float* ln1_g   = (const float*)d_in[7];
    const float* ln1_b   = (const float*)d_in[8];
    const float* ln2_g   = (const float*)d_in[9];
    const float* ln2_b   = (const float*)d_in[10];
    const float* grw     = (const float*)d_in[11];
    const float* grb     = (const float*)d_in[12];
    const float* erw     = (const float*)d_in[13];
    const float* erb     = (const float*)d_in[14];
    const float* w1      = (const float*)d_in[15];
    const float* b1      = (const float*)d_in[16];
    const float* w2      = (const float*)d_in[17];
    const float* b2      = (const float*)d_in[18];
    const float* lnf_g   = (const float*)d_in[19];
    const float* lnf_b   = (const float*)d_in[20];
    const float* head_w  = (const float*)d_in[21];
    const float* head_b  = (const float*)d_in[22];
    float* out = (float*)d_out;

    // -------- workspace carve (~118 MiB; all chunks 16B-aligned) --------
    char* w = (char*)d_ws;
    float*  h     = (float*)w;  w += (size_t)N_ * D_ * 4;             // 4 MiB
    float*  xln   = (float*)w;  w += (size_t)N_ * D_ * 4;             // 4 MiB (router)
    float*  qkvb  = (float*)w;  w += (size_t)N_ * 3 * D_ * 4;         // 12 MiB
    float*  gate  = (float*)w;  w += (size_t)N_ * 4;
    int*    counts= (int*)w;    w += 16 * 4;
    int*    idxbuf= (int*)w;    w += (size_t)NEXP * N_ * 4;
    __bf16* xlnb  = (__bf16*)w; w += (size_t)N_ * D_ * 2;             // 2 MiB
    __bf16* attnb = (__bf16*)w; w += (size_t)N_ * D_ * 2;             // 2 MiB
    __bf16* h1b   = (__bf16*)w; w += (size_t)N_ * H_ * 2;             // 4 MiB
    __bf16* wqkvb = (__bf16*)w; w += (size_t)L_ * 3 * D_ * D_ * 2;    // 3 MiB
    __bf16* wob   = (__bf16*)w; w += (size_t)L_ * D_ * D_ * 2;        // 1 MiB
    __bf16* w1t   = (__bf16*)w; w += (size_t)L_ * NEXP * D_ * H_ * 2; // 16 MiB
    __bf16* w2t   = (__bf16*)w; w += (size_t)L_ * NEXP * H_ * D_ * 2; // 16 MiB
    __bf16* headb = (__bf16*)w; w += (size_t)V_ * D_ * 2;             // 51.5 MiB

    // -------- weight prep: convert (+transpose) to bf16, K-contiguous --------
    {
        long n;
        n = (long)L_ * 3 * D_ * D_;
        cvt_bf16_kernel<<<(n + 255) / 256, 256, 0, stream>>>(wqkv, wqkvb, n);
        n = (long)L_ * D_ * D_;
        cvt_bf16_kernel<<<(n + 255) / 256, 256, 0, stream>>>(wo, wob, n);
        n = (long)V_ * D_;
        cvt_bf16_kernel<<<(n + 255) / 256, 256, 0, stream>>>(head_w, headb, n);
        // w1 [l,e][D,H] -> w1t [l,e][H,D] ; w2 [l,e][H,D] -> w2t [l,e][D,H]
        transpose_cvt_kernel<<<dim3((D_ * H_) / 256, 1, L_ * NEXP), 256, 0, stream>>>(
            w1, w1t, D_, H_);
        transpose_cvt_kernel<<<dim3((H_ * D_) / 256, 1, L_ * NEXP), 256, 0, stream>>>(
            w2, w2t, H_, D_);
    }

    embed_kernel<<<(N_ * D_) / 256, 256, 0, stream>>>(tokens, tok_emb, pos_emb, h);

    for (int l = 0; l < L_; l++) {
        // --- attention block ---
        layernorm<<<N_ / 8, 256, 0, stream>>>(h, ln1_g + l * D_, ln1_b + l * D_, xln, xlnb);
        wmma_gemm_bf<<<dim3(N_ / 16, 3), 256, 0, stream>>>(
            xlnb, D_, wqkvb + (size_t)l * 3 * D_ * D_,
            bqkv + (size_t)l * 3 * D_, qkvb, 3 * D_, N_, 3 * D_, D_, 0);
        attn_kernel<<<(B_ * NH_ * S_) / 8, 256, 0, stream>>>(qkvb, attnb);
        wmma_gemm_bf<<<dim3(N_ / 16, 1), 256, 0, stream>>>(
            attnb, D_, wob + (size_t)l * D_ * D_,
            bo + (size_t)l * D_, h, D_, N_, D_, D_, 1);

        // --- MoE block ---
        layernorm<<<N_ / 8, 256, 0, stream>>>(h, ln2_g + l * D_, ln2_b + l * D_, xln, xlnb);
        (void)hipMemsetAsync(counts, 0, NEXP * sizeof(int), stream);
        moe_router<<<N_ / 8, 256, 0, stream>>>(
            xln, grw + (size_t)l * G_ * D_, grb + (size_t)l * G_,
            erw + (size_t)l * G_ * E_ * D_, erb + (size_t)l * G_ * E_,
            counts, idxbuf, gate);
        // FFN1: h1b = bf16(relu(x @ w1 + b1)); B rows = w1t [H,D]
        moe_gemm_bf<<<dim3(N_ / 16, H_ / (16 * TN * 8), NEXP), 256, 0, stream>>>(
            xlnb, D_, w1t + (size_t)l * NEXP * H_ * D_, (long)H_ * D_,
            b1 + (size_t)l * NEXP * H_, H_, h1b, (float*)nullptr, H_,
            counts, idxbuf, gate, H_, D_, 0);
        // FFN2: h += gate * (h1 @ w2 + b2); B rows = w2t [D,H]
        moe_gemm_bf<<<dim3(N_ / 16, D_ / (16 * TN * 8), NEXP), 256, 0, stream>>>(
            h1b, H_, w2t + (size_t)l * NEXP * D_ * H_, (long)D_ * H_,
            b2 + (size_t)l * NEXP * D_, D_, (__bf16*)nullptr, h, D_,
            counts, idxbuf, gate, D_, H_, 1);
    }

    // --- final LN + LM head ---
    layernorm<<<N_ / 8, 256, 0, stream>>>(h, lnf_g, lnf_b, xln, xlnb);
    wmma_gemm_bf<<<dim3(N_ / 16, (V_ + 16 * TN * 8 - 1) / (16 * TN * 8)), 256, 0, stream>>>(
        xlnb, D_, headb, head_b, out, V_, N_, V_, D_, 0);
}
